// LongformerClassifier_24833500905612
// MI455X (gfx1250) — matrix operand order repaired and checked by reading, not compiled, not run on hardware
//
#include <hip/hip_runtime.h>
#include <cstdint>

#define S_LEN  4096
#define NHEAD  12
#define DHEAD  64
#define DMODEL 768
#define BATCH  2
#define WWIN   256
#define NGLOB  8
#define NEGV   (-1e9f)

typedef __attribute__((ext_vector_type(16))) __bf16 bf16x16;
typedef __attribute__((ext_vector_type(8)))  float  f32x8;

union ABFrag { bf16x16 v; uint32_t d[8]; uint16_t e[16]; };

__device__ inline uint16_t f2bf(float f) {
  uint32_t u = __float_as_uint(f);
  u += 0x7FFFu + ((u >> 16) & 1u);          // round-to-nearest-even
  return (uint16_t)(u >> 16);
}

__device__ inline f32x8 wmma_bf16(const ABFrag& a, const ABFrag& b, f32x8 c) {
  return __builtin_amdgcn_wmma_f32_16x16x32_bf16(false, a.v, false, b.v,
                                                 (short)0, c, false, false);
}

// Async global->LDS copy, 16B per lane. LDS dest address comes from the low 32
// bits of the generic pointer (flat->LDS truncation per ISA 10.2). Tracked by
// ASYNCcnt; must s_wait_asynccnt before consuming.
__device__ inline void async_g2l_b128(void* lds, const void* gptr) {
  uint32_t l = (uint32_t)(uintptr_t)lds;
  uint64_t g = (uint64_t)(uintptr_t)gptr;
  asm volatile("global_load_async_to_lds_b128 %0, %1, off"
               :: "v"(l), "v"(g) : "memory");
}
__device__ inline void wait_async0() {
  asm volatile("s_wait_asynccnt 0x0" ::: "memory");
}

// A-fragment 16x32 bf16: lanes 0-15 row M=l15 K=0..15, lanes 16-31 K=16..31
__device__ inline void load_a(ABFrag& f, const uint32_t* base, int row, int rowDw,
                              int kDw, int hi) {
  int o = row * rowDw + kDw + 8 * hi;
#pragma unroll
  for (int j = 0; j < 8; j++) f.d[j] = base[o + j];
}

// B-fragment 32x16 bf16 from K-contiguous layout base[n][k]:
// lanes 0-15 col N=l15 K=0..15, lanes 16-31 col N=l15 K=16..31
__device__ inline void load_b(ABFrag& f, const uint32_t* base, int ncol, int rowDw,
                              int kDw, int hi) {
  int o = ncol * rowDw + kDw + 8 * hi;
#pragma unroll
  for (int j = 0; j < 8; j++) f.d[j] = base[o + j];
}

__device__ inline f32x8 zero8() {
  f32x8 z;
#pragma unroll
  for (int i = 0; i < 8; i++) z[i] = 0.0f;
  return z;
}

// ---------------------------------------------------------------- convert
__global__ __launch_bounds__(256) void cvt_bf16(const float* __restrict__ in,
                                                uint16_t* __restrict__ out, int n) {
  int i = blockIdx.x * 256 + threadIdx.x;
  if (i < n) out[i] = f2bf(in[i]);
}

// ---------------------------------------------------------------- GEMM X(8192x768) @ W(768x768)
// Output written head-major bf16: dst[((b*H+h)*S + s)*64 + dh], optional scale.
__global__ __launch_bounds__(256) void gemm768(const uint16_t* __restrict__ X,
                                               const uint16_t* __restrict__ W,
                                               uint16_t* __restrict__ dst, float scale) {
  __shared__ uint16_t aS[128 * 32];   // A tile  [row][k]
  __shared__ uint16_t btS[64 * 32];   // B tile transposed [n][k]
  const int m0 = blockIdx.x * 128;
  const int n0 = blockIdx.y * 64;
  const int tid = threadIdx.x;
  const int lane = tid & 31;
  const int wv = tid >> 5;
  const int hi = lane >> 4;
  const int l15 = lane & 15;

  f32x8 acc[4];
#pragma unroll
  for (int nt = 0; nt < 4; nt++) acc[nt] = zero8();

  for (int k0 = 0; k0 < DMODEL; k0 += 32) {
    __syncthreads();
    {   // stage A: 128x32 bf16 via async DMA (32B/thread, contiguous)
      int r = tid >> 1, c = (tid & 1) * 16;
      const uint16_t* src = X + (size_t)(m0 + r) * DMODEL + k0 + c;
      async_g2l_b128(aS + r * 32 + c,     src);
      async_g2l_b128(aS + r * 32 + c + 8, src + 8);
      // stage B transposed: 32x64 bf16, 8 elems/thread (16-bit scatter)
      int kr = tid >> 3, nc = (tid & 7) * 8;
      uint4 wv4 = *(const uint4*)(W + (size_t)(k0 + kr) * DMODEL + n0 + nc);
      const uint16_t* wp = (const uint16_t*)&wv4;
#pragma unroll
      for (int j = 0; j < 8; j++) btS[(nc + j) * 32 + kr] = wp[j];
      if (k0 + 32 < DMODEL) {   // prefetch next tiles
        __builtin_prefetch(src + 32, 0, 1);
        __builtin_prefetch(W + (size_t)(k0 + 32 + kr) * DMODEL + n0 + nc, 0, 1);
      }
    }
    wait_async0();
    __syncthreads();

    const uint32_t* a32 = (const uint32_t*)aS;
    const uint32_t* b32 = (const uint32_t*)btS;
    ABFrag a;
    load_a(a, a32, wv * 16 + l15, 16, 0, hi);
#pragma unroll
    for (int nt = 0; nt < 4; nt++) {
      ABFrag b;
      load_b(b, b32, nt * 16 + l15, 16, 0, hi);
      acc[nt] = wmma_bf16(a, b, acc[nt]);
    }
  }

  const int h = n0 >> 6;
#pragma unroll
  for (int nt = 0; nt < 4; nt++) {
#pragma unroll
    for (int r = 0; r < 8; r++) {
      int m = m0 + wv * 16 + r + 8 * hi;
      int b = m >> 12, s = m & (S_LEN - 1);
      int dh = nt * 16 + l15;
      dst[((size_t)(b * NHEAD + h) * S_LEN + s) * DHEAD + dh] = f2bf(acc[nt][r] * scale);
    }
  }
}

// ---------------------------------------------------------------- windowed attention
// One workgroup per (chunk n, head h, batch b); 8 waves, each owns 32 query rows.
__global__ __launch_bounds__(256) void attn_win(const uint16_t* __restrict__ qbf,
                                                const uint16_t* __restrict__ kbf,
                                                const uint16_t* __restrict__ vbf,
                                                const float* __restrict__ amask,
                                                float* __restrict__ out) {
  __shared__ uint16_t kS[32 * 64];        // [key][dh]
  __shared__ uint16_t vS[64 * 32];        // transposed [dh][key]
  __shared__ uint16_t pS[8][32 * 32];     // per-wave P transpose buffer
  const int n = blockIdx.x, h = blockIdx.y, b = blockIdx.z;
  const int tid = threadIdx.x, lane = tid & 31, wv = tid >> 5;
  const int hi = lane >> 4, l15 = lane & 15;

  const size_t headbase = ((size_t)(b * NHEAD) + h) * S_LEN * DHEAD;
  const uint32_t* q32 = (const uint32_t*)(qbf + headbase);
  const uint16_t* kB = kbf + headbase;
  const uint16_t* vB = vbf + headbase;

  ABFrag aQ[2][2];
#pragma unroll
  for (int t = 0; t < 2; t++) {
    int row = n * WWIN + wv * 32 + t * 16 + l15;
#pragma unroll
    for (int kh = 0; kh < 2; kh++) load_a(aQ[t][kh], q32, row, 32, 16 * kh, hi);
  }

  f32x8 acc[2][4];
  float mrun[2][8], lrun[2][8];
#pragma unroll
  for (int t = 0; t < 2; t++) {
#pragma unroll
    for (int nt = 0; nt < 4; nt++) acc[t][nt] = zero8();
#pragma unroll
    for (int r = 0; r < 8; r++) { mrun[t][r] = -3e38f; lrun[t][r] = 0.0f; }
  }

  for (int blk = -1; blk < 24; blk++) {
    const bool isglob = (blk < 0);
    const int kpbase = isglob ? 0 : (n * WWIN - WWIN + blk * 32);
    __syncthreads();
    {   // cooperative staging of 32 keys of K (async DMA) and V (transposed)
      int key = tid >> 3, dh = (tid & 7) * 8;
      int kp = kpbase + key;
      int kc = min(max(kp, 0), S_LEN - 1);
      async_g2l_b128(kS + key * DHEAD + dh, kB + (size_t)kc * DHEAD + dh);
      uint4 y = *(const uint4*)(vB + (size_t)kc * DHEAD + dh);
      const uint16_t* yp = (const uint16_t*)&y;
#pragma unroll
      for (int j = 0; j < 8; j++) vS[(dh + j) * 32 + key] = yp[j];
      // prefetch next key block
      int kpn = min(max(kpbase + 32 + key, 0), S_LEN - 1);
      __builtin_prefetch(kB + (size_t)kpn * DHEAD + dh, 0, 1);
      __builtin_prefetch(vB + (size_t)kpn * DHEAD + dh, 0, 1);
    }
    wait_async0();
    __syncthreads();

    const uint32_t* k32 = (const uint32_t*)kS;
    const uint32_t* v32 = (const uint32_t*)vS;

#pragma unroll
    for (int t = 0; t < 2; t++) {
      f32x8 sc[2];
#pragma unroll
      for (int ct = 0; ct < 2; ct++) {
        f32x8 s = zero8();
        ABFrag bb;
#pragma unroll
        for (int kh = 0; kh < 2; kh++) {
          load_b(bb, k32, ct * 16 + l15, 32, 16 * kh, hi);
          s = wmma_bf16(aQ[t][kh], bb, s);
        }
        int col = ct * 16 + l15;
        int kp = kpbase + col;
        bool cok;
        if (isglob) cok = (col < NGLOB);
        else cok = (kp >= NGLOB) && (kp < S_LEN) &&
                   (amask[b * S_LEN + min(max(kp, 0), S_LEN - 1)] > 0.5f);
#pragma unroll
        for (int r = 0; r < 8; r++) {
          int qpos = n * WWIN + wv * 32 + t * 16 + r + 8 * hi;
          bool ok = cok && (isglob || ((kp >= qpos - WWIN) && (kp <= qpos + WWIN)));
          s[r] = ok ? s[r] : NEGV;
        }
        sc[ct] = s;
      }
      // online softmax update for this M-tile
#pragma unroll
      for (int r = 0; r < 8; r++) {
        float mx = fmaxf(sc[0][r], sc[1][r]);
#pragma unroll
        for (int msk = 1; msk < 16; msk <<= 1) mx = fmaxf(mx, __shfl_xor(mx, msk));
        float mnew = fmaxf(mrun[t][r], mx);
        float alpha = __expf(mrun[t][r] - mnew);
        mrun[t][r] = mnew;
        float p0 = (sc[0][r] < -1e8f) ? 0.0f : __expf(sc[0][r] - mnew);
        float p1 = (sc[1][r] < -1e8f) ? 0.0f : __expf(sc[1][r] - mnew);
        float rs = p0 + p1;
#pragma unroll
        for (int msk = 1; msk < 16; msk <<= 1) rs += __shfl_xor(rs, msk);
        lrun[t][r] = lrun[t][r] * alpha + rs;
#pragma unroll
        for (int nt = 0; nt < 4; nt++) acc[t][nt][r] *= alpha;
        int prow = t * 16 + r + 8 * hi;
        pS[wv][prow * 32 + l15]      = f2bf(p0);
        pS[wv][prow * 32 + 16 + l15] = f2bf(p1);
      }
    }
    __syncthreads();   // make P visible (also keeps waves in lockstep for staging)

    const uint32_t* p32 = (const uint32_t*)pS[wv];
#pragma unroll
    for (int t = 0; t < 2; t++) {
      ABFrag ap;
      load_a(ap, p32, t * 16 + l15, 16, 0, hi);
#pragma unroll
      for (int nt = 0; nt < 4; nt++) {
        ABFrag bv;
        load_b(bv, v32, nt * 16 + l15, 16, 0, hi);
        acc[t][nt] = wmma_bf16(ap, bv, acc[t][nt]);
      }
    }
  }

#pragma unroll
  for (int t = 0; t < 2; t++) {
#pragma unroll
    for (int r = 0; r < 8; r++) {
      float inv = 1.0f / lrun[t][r];
      int s = n * WWIN + wv * 32 + t * 16 + r + 8 * hi;
#pragma unroll
      for (int nt = 0; nt < 4; nt++) {
        int dh = nt * 16 + l15;
        out[((size_t)b * S_LEN + s) * DMODEL + h * DHEAD + dh] = acc[t][nt][r] * inv;
      }
    }
  }
}

// ---------------------------------------------------------------- global attention rows
// One wave per (b,h): 8 global queries (padded to 16) vs all S keys. Overwrites rows 0..7.
__global__ __launch_bounds__(32) void attn_glob(const uint16_t* __restrict__ qgbf,
                                                const uint16_t* __restrict__ kgbf,
                                                const uint16_t* __restrict__ vgbf,
                                                const float* __restrict__ amask,
                                                float* __restrict__ out) {
  const int h = blockIdx.x % NHEAD, b = blockIdx.x / NHEAD;
  const int lane = threadIdx.x & 31, hi = lane >> 4, l15 = lane & 15;
  __shared__ uint16_t pS[16 * 32];

  const size_t headbase = ((size_t)(b * NHEAD) + h) * S_LEN * DHEAD;
  const uint32_t* q32 = (const uint32_t*)(qgbf + headbase);
  const uint32_t* kg32 = (const uint32_t*)(kgbf + headbase);
  const uint16_t* vg = vgbf + headbase;

  ABFrag aQ[2];
#pragma unroll
  for (int kh = 0; kh < 2; kh++) load_a(aQ[kh], q32, l15, 32, 16 * kh, hi);

  f32x8 acc[4];
  float mrun[8], lrun[8];
#pragma unroll
  for (int nt = 0; nt < 4; nt++) acc[nt] = zero8();
#pragma unroll
  for (int r = 0; r < 8; r++) { mrun[r] = -3e38f; lrun[r] = 0.0f; }

  for (int kb = 0; kb < S_LEN; kb += 32) {
    __builtin_prefetch(vg + (size_t)(kb + 32 + lane) * DHEAD, 0, 1);
    f32x8 sc[2];
#pragma unroll
    for (int ct = 0; ct < 2; ct++) {
      f32x8 s = zero8();
      ABFrag bb;
      int key = kb + ct * 16 + l15;
#pragma unroll
      for (int kh = 0; kh < 2; kh++) {
        load_b(bb, kg32, key, 32, 16 * kh, hi);
        s = wmma_bf16(aQ[kh], bb, s);
      }
      bool ok = amask[b * S_LEN + key] > 0.5f;
#pragma unroll
      for (int r = 0; r < 8; r++) s[r] = ok ? s[r] : NEGV;
      sc[ct] = s;
    }
#pragma unroll
    for (int r = 0; r < 8; r++) {
      float mx = fmaxf(sc[0][r], sc[1][r]);
#pragma unroll
      for (int msk = 1; msk < 16; msk <<= 1) mx = fmaxf(mx, __shfl_xor(mx, msk));
      float mnew = fmaxf(mrun[r], mx);
      float alpha = __expf(mrun[r] - mnew);
      mrun[r] = mnew;
      float p0 = (sc[0][r] < -1e8f) ? 0.0f : __expf(sc[0][r] - mnew);
      float p1 = (sc[1][r] < -1e8f) ? 0.0f : __expf(sc[1][r] - mnew);
      float rs = p0 + p1;
#pragma unroll
      for (int msk = 1; msk < 16; msk <<= 1) rs += __shfl_xor(rs, msk);
      lrun[r] = lrun[r] * alpha + rs;
#pragma unroll
      for (int nt = 0; nt < 4; nt++) acc[nt][r] *= alpha;
      int prow = r + 8 * hi;
      pS[prow * 32 + l15]      = f2bf(p0);
      pS[prow * 32 + 16 + l15] = f2bf(p1);
    }
    __syncthreads();

    ABFrag ap;
    load_a(ap, (const uint32_t*)pS, l15, 16, 0, hi);
#pragma unroll
    for (int nt = 0; nt < 4; nt++) {
      ABFrag bv;
      int dh = nt * 16 + l15;
#pragma unroll
      for (int j = 0; j < 16; j++)
        bv.e[j] = vg[(size_t)(kb + 16 * hi + j) * DHEAD + dh];
      acc[nt] = wmma_bf16(ap, bv, acc[nt]);
    }
    __syncthreads();
  }

  if (hi == 0) {   // only rows 0..7 are real global queries
#pragma unroll
    for (int r = 0; r < 8; r++) {
      float inv = 1.0f / lrun[r];
#pragma unroll
      for (int nt = 0; nt < 4; nt++) {
        int dh = nt * 16 + l15;
        out[((size_t)b * S_LEN + r) * DMODEL + h * DHEAD + dh] = acc[nt][r] * inv;
      }
    }
  }
}

// ---------------------------------------------------------------- host
extern "C" void kernel_launch(void* const* d_in, const int* in_sizes, int n_in,
                              void* d_out, int out_size, void* d_ws, size_t ws_size,
                              hipStream_t stream) {
  (void)in_sizes; (void)n_in; (void)out_size; (void)ws_size;
  const float* hs = (const float*)d_in[0];
  const float* am = (const float*)d_in[1];
  const float* Wmat[6] = { (const float*)d_in[2], (const float*)d_in[3],
                           (const float*)d_in[4], (const float*)d_in[5],
                           (const float*)d_in[6], (const float*)d_in[7] };

  uint8_t* ws = (uint8_t*)d_ws;
  size_t off = 0;
  auto alloc = [&](size_t bytes) -> uint16_t* {
    uint16_t* p = (uint16_t*)(ws + off);
    off = (off + bytes + 255) & ~(size_t)255;
    return p;
  };
  const size_t xElems = (size_t)BATCH * S_LEN * DMODEL;        // 6,291,456
  const size_t wElems = (size_t)DMODEL * DMODEL;               // 589,824
  const size_t pElems = (size_t)BATCH * NHEAD * S_LEN * DHEAD; // 6,291,456

  uint16_t* Xbf = alloc(xElems * 2);
  uint16_t* Wbf[6];
  for (int i = 0; i < 6; i++) Wbf[i] = alloc(wElems * 2);
  uint16_t* proj[6];
  for (int i = 0; i < 6; i++) proj[i] = alloc(pElems * 2);   // q,k,v,qg,kg,vg

  cvt_bf16<<<(int)((xElems + 255) / 256), 256, 0, stream>>>(hs, Xbf, (int)xElems);
  for (int i = 0; i < 6; i++)
    cvt_bf16<<<(int)((wElems + 255) / 256), 256, 0, stream>>>(Wmat[i], Wbf[i], (int)wElems);

  const float scale = 0.125f;   // 1/sqrt(64)
  const float scl[6] = { scale, 1.0f, 1.0f, scale, 1.0f, 1.0f };
  dim3 gg(64, 12);              // 8192/128 x 768/64
  for (int i = 0; i < 6; i++)
    gemm768<<<gg, 256, 0, stream>>>(Xbf, Wbf[i], proj[i], scl[i]);

  dim3 ga(S_LEN / WWIN, NHEAD, BATCH);   // (16,12,2)
  attn_win<<<ga, 256, 0, stream>>>(proj[0], proj[1], proj[2], am, (float*)d_out);
  attn_glob<<<BATCH * NHEAD, 32, 0, stream>>>(proj[3], proj[4], proj[5], am,
                                              (float*)d_out);
}
